// Transducer_28166395527675
// MI455X (gfx1250) — compile-verified, matrix operand order checked by reading, not run on hardware
//
#include <hip/hip_runtime.h>
#include <math.h>

// ---------------------------------------------------------------------------
// Problem constants (match reference setup_inputs)
// ---------------------------------------------------------------------------
#define N_  4
#define T_  256
#define U_  64
#define U1_ 65          // U + 1
#define C_  1024
#define R_  (N_ * T_ * U1_)          // 66,560 rows
#define NEG (-1.0e9f)

// reverse-label-smoothing constants (alpha = 0.1, C = 1024)
#define ONE_MINUS_A 0.9f
#define A_OVER_C    9.765625e-05f
#define TGT_ENTROPY 1.0182301539513936f   // -((1-a)ln(1-a) + a ln(a/C))
#define LOG2E       1.44269504088896340736f

typedef __attribute__((ext_vector_type(16))) _Float16 v16h;
typedef __attribute__((ext_vector_type(8)))  float    v8f;
typedef __attribute__((ext_vector_type(4)))  float    f4;

__device__ __forceinline__ float sel8(const v8f v, int i) {
    float r = v[0];
    r = (i == 1) ? v[1] : r;
    r = (i == 2) ? v[2] : r;
    r = (i == 3) ? v[3] : r;
    r = (i == 4) ? v[4] : r;
    r = (i == 5) ? v[5] : r;
    r = (i == 6) ? v[6] : r;
    r = (i == 7) ? v[7] : r;
    return r;
}

__device__ __forceinline__ float sel8a(const float* v, int i) {
    float r = v[0];
    r = (i == 1) ? v[1] : r;
    r = (i == 2) ? v[2] : r;
    r = (i == 3) ? v[3] : r;
    r = (i == 4) ? v[4] : r;
    r = (i == 5) ? v[5] : r;
    r = (i == 6) ? v[6] : r;
    r = (i == 7) ? v[7] : r;
    return r;
}

// ---------------------------------------------------------------------------
// Kernel 1: streaming row reduction.
// One wave handles 16 rows of C=1024. Per K-tile of 32 columns:
//   lane L<16  : row rb+L, cols c..c+7  and c+16..c+23   (A-matrix K=0..7,16..23)
//   lane L>=16 : row rb+L-16, cols c+8..c+15, c+24..c+31 (A-matrix K=8..15,24..31)
// Sum(x) over C is accumulated with v_wmma_f32_16x16x32_f16 against an all-ones
// B matrix (fp32 accumulator). max / sum(exp) use TILE-level online softmax:
// per 16-element tile: tree-max -> 16 independent exps -> tree-sum -> one fma
// rescale of the running sum. exp(x-m) is computed as
// exp2(fma(x, log2e, -m*log2e)) : one v_fma + one native v_exp_f32 per element.
// ---------------------------------------------------------------------------
__global__ void __launch_bounds__(256)
rnnt_row_reduce(const float* __restrict__ logits,
                const int*   __restrict__ targets,
                float* __restrict__ blankS,
                float* __restrict__ emitS)
{
    const int lane = threadIdx.x & 31;
    const int wv   = blockIdx.x * 8 + (threadIdx.x >> 5);
    const int rb   = wv * 16;                 // first of 16 rows for this wave
    const int lrow = lane & 15;
    const int row  = rb + lrow;
    const bool hi  = lane >= 16;

    // row -> (n, t, u) ; only n,u needed for the target id
    const int n   = row / (T_ * U1_);
    const int rem = row - n * (T_ * U1_);
    const int u   = rem % U1_;
    const int tgt = (u < U_) ? targets[n * U_ + u] : 0;

    const float* __restrict__ rp = logits + (size_t)row * C_;

    v16h bones;
#pragma unroll
    for (int i = 0; i < 16; ++i) bones[i] = (_Float16)1.0f;

    v8f   acc = {};                  // fp32 WMMA accumulator: row sums of x
    float m = -INFINITY, s = 0.0f;   // running (max, sum-exp) for lane subset

#pragma unroll 2
    for (int it = 0; it < 32; ++it) {
        const int c0 = it * 32 + (hi ? 8 : 0);
        __builtin_prefetch(rp + c0 + 64, 0, 1);      // global_prefetch ahead

        const f4 q0 = __builtin_nontemporal_load((const f4*)(rp + c0));
        const f4 q1 = __builtin_nontemporal_load((const f4*)(rp + c0 + 4));
        const f4 q2 = __builtin_nontemporal_load((const f4*)(rp + c0 + 16));
        const f4 q3 = __builtin_nontemporal_load((const f4*)(rp + c0 + 20));

        float f[16];
#pragma unroll
        for (int j = 0; j < 4; ++j) { f[j] = q0[j]; f[4+j] = q1[j];
                                      f[8+j] = q2[j]; f[12+j] = q3[j]; }

        v16h a;
#pragma unroll
        for (int j = 0; j < 16; ++j) a[j] = (_Float16)f[j];

        // ---- tile max (4-level tree, fully parallel; compiler fuses max3) ----
        float t8[8], t4[4], t2[2];
#pragma unroll
        for (int j = 0; j < 8; ++j) t8[j] = fmaxf(f[j], f[j + 8]);
#pragma unroll
        for (int j = 0; j < 4; ++j) t4[j] = fmaxf(t8[j], t8[j + 4]);
        t2[0] = fmaxf(t4[0], t4[2]);
        t2[1] = fmaxf(t4[1], t4[3]);
        const float tmax = fmaxf(t2[0], t2[1]);
        const float mnew = fmaxf(m, tmax);

        // ---- 16 independent exps (1 fma + 1 v_exp each) + tree sum ----
        const float nl2m = -LOG2E * mnew;    // hoisted: one mul per tile
        float e[16];
#pragma unroll
        for (int j = 0; j < 16; ++j)
            e[j] = __builtin_amdgcn_exp2f(fmaf(f[j], LOG2E, nl2m));
        float s8[8], s4[4], s2[2];
#pragma unroll
        for (int j = 0; j < 8; ++j) s8[j] = e[j] + e[j + 8];
#pragma unroll
        for (int j = 0; j < 4; ++j) s4[j] = s8[j] + s8[j + 4];
        s2[0] = s4[0] + s4[2];
        s2[1] = s4[1] + s4[3];
        const float esum = s2[0] + s2[1];

        // rescale running sum (first iter: exp2(-inf)=0 -> s = esum)
        s = fmaf(s, __builtin_amdgcn_exp2f((m - mnew) * LOG2E), esum);
        m = mnew;

        // D = A * ones + D : accumulates per-row sum of x across K (fp32 acc)
        acc = __builtin_amdgcn_wmma_f32_16x16x32_f16(
                  false, a, false, bones, (short)0, acc, false, false);
    }

    // merge the lane pair (L, L+16) that shared each row
    const float m2 = __shfl_xor(m, 16, 32);
    const float s2 = __shfl_xor(s, 16, 32);
    const float M  = fmaxf(m, m2);
    const float S  = s  * __builtin_amdgcn_exp2f((m  - M) * LOG2E)
                   + s2 * __builtin_amdgcn_exp2f((m2 - M) * LOG2E);

    // extract this row's Sum(x) from the D-matrix layout:
    //   vgpr j, lanes 0-15 -> M=j ; lanes 16-31 -> M=j+8 (all N columns equal)
    float accx[8];
#pragma unroll
    for (int j = 0; j < 8; ++j) accx[j] = __shfl_xor(acc[j], 16, 32);
    const float lo   = sel8(acc,  lane & 7);
    const float hv   = sel8a(accx, lane & 7);
    const float sumx = (lane < 8) ? lo : hv;

    if (lane < 16) {
        // blank / emit logits: two tiny reloads instead of per-element capture
        const float bx = rp[0];
        const float ex = rp[tgt];
        const float lse  = M + __logf(S);
        const float slp  = sumx - (float)C_ * lse;       // sum of log-probs
        const float base = slp * A_OVER_C + TGT_ENTROPY;
        blankS[row] = (bx - lse) * ONE_MINUS_A + base;
        emitS[row]  = (ex - lse) * ONE_MINUS_A + base;
    }
}

// ---------------------------------------------------------------------------
// Kernel 2: anti-diagonal wavefront DP (log-space forward algorithm).
// alpha[t,u] = lae(alpha[t-1,u] + blank[t-1,u], alpha[t,u-1] + emit[t,u-1])
// All cells on diagonal d = t+u depend only on diagonal d-1 -> 65-wide parallel.
// ---------------------------------------------------------------------------
__device__ __forceinline__ float lae(float a, float b) {
    const float mx = fmaxf(a, b);
    const float d  = fabsf(a - b);
    return mx + log1pf(__builtin_amdgcn_exp2f(-d * LOG2E));
}

__global__ void __launch_bounds__(128)
rnnt_dp(const float* __restrict__ blankS,
        const float* __restrict__ emitS,
        const int*   __restrict__ xlens,
        const int*   __restrict__ ylens,
        float* __restrict__ out)
{
    __shared__ float bufA[U1_];
    __shared__ float bufB[U1_];
    __shared__ float s_term;
    __shared__ float s_total;

    const int tid = threadIdx.x;
    if (tid == 0) s_total = 0.0f;

    for (int n = 0; n < N_; ++n) {
        const int xl = xlens[n];
        const int yl = ylens[n];
        float* Ap = bufA;
        float* Ac = bufB;
        __syncthreads();

        for (int d = 0; d < T_ + U1_ - 1; ++d) {
            const int uu = tid;
            const int t  = d - uu;
            const bool active = (uu < U1_) && (t >= 0) && (t < T_);
            float val = NEG;
            if (active) {
                const float left = (t == 0)
                    ? ((uu == 0) ? 0.0f : NEG)
                    : Ap[uu] + blankS[(size_t)(n * T_ + (t - 1)) * U1_ + uu];
                const float down = (uu == 0)
                    ? NEG
                    : Ap[uu - 1] + emitS[(size_t)(n * T_ + t) * U1_ + (uu - 1)];
                val = lae(left, down);
                if (t == xl - 1 && uu == yl)
                    s_term = val + blankS[(size_t)(n * T_ + t) * U1_ + uu];
            }
            if (active) Ac[uu] = val;
            __syncthreads();
            float* tp = Ap; Ap = Ac; Ac = tp;
        }

        __syncthreads();
        if (tid == 0) s_total += s_term;
        __syncthreads();
    }

    if (tid == 0) out[0] = -s_total;
}

// ---------------------------------------------------------------------------
// Launch
// ---------------------------------------------------------------------------
extern "C" void kernel_launch(void* const* d_in, const int* in_sizes, int n_in,
                              void* d_out, int out_size, void* d_ws, size_t ws_size,
                              hipStream_t stream)
{
    const float* logits  = (const float*)d_in[0];
    const int*   targets = (const int*)d_in[1];
    const int*   xlens   = (const int*)d_in[2];
    const int*   ylens   = (const int*)d_in[3];
    float*       out     = (float*)d_out;

    float* blankS = (float*)d_ws;         // R_ floats
    float* emitS  = blankS + R_;          // R_ floats  (~532 KB total)

    // 66,560 rows / 16 rows-per-wave = 4160 waves = 520 blocks * 8 waves
    rnnt_row_reduce<<<520, 256, 0, stream>>>(logits, targets, blankS, emitS);
    rnnt_dp<<<1, 128, 0, stream>>>(blankS, emitS, xlens, ylens, out);
}
